// GCN2_35510789603589
// MI455X (gfx1250) — matrix operand order repaired and checked by reading, not compiled, not run on hardware
//
#include <hip/hip_runtime.h>

typedef _Float16 f16;
typedef __attribute__((ext_vector_type(16))) _Float16 v16h;
typedef __attribute__((ext_vector_type(8)))  _Float16 v8h;
typedef __attribute__((ext_vector_type(8)))  float    v8f;

#define NNODES   50000
#define DEG      32
#define F        128     // NUM_FEATS == NUM_HIDDEN
#define KDIM     256     // 2*F
#define NLAB     40
#define NLABP    48      // padded to 3 N-tiles of 16
#define ZSTR     264     // LDS row stride in halves (256 + 8 pad -> conflict-free)

// ---------------------------------------------------------------------------
// Prep: transpose W1 [256,128]->w1t[128][256] f16, W2 [256,40]->w2t[48][256] f16
// ---------------------------------------------------------------------------
__global__ void gcn_prep_weights(const float* __restrict__ W1,
                                 const float* __restrict__ W2,
                                 f16* __restrict__ w1t, f16* __restrict__ w2t) {
  int i = blockIdx.x * blockDim.x + threadIdx.x;
  if (i < F * KDIM) {                       // 32768 entries
    int n = i / KDIM, k = i % KDIM;
    w1t[i] = (f16)W1[k * F + n];
  }
  int j = i - F * KDIM;
  if (j >= 0 && j < NLABP * KDIM) {         // 12288 entries
    int n = j / KDIM, k = j % KDIM;
    w2t[j] = (n < NLAB) ? (f16)W2[k * NLAB + n] : (f16)0.0f;
  }
}

// ---------------------------------------------------------------------------
// Layer 1: z1 = [feats | mean(neigh feats)]  (16 nodes / block, LDS f16 tile)
//          h1 = relu(z1 @ W1 + b1)  via v_wmma_f32_16x16x32_f16, stored f16
// ---------------------------------------------------------------------------
__global__ __launch_bounds__(128)
void gcn_layer1(const float* __restrict__ feats, const int* __restrict__ neigh,
                const f16* __restrict__ w1t, const float* __restrict__ b1,
                f16* __restrict__ h1) {
  __shared__ f16 z[16 * ZSTR];
  const int t     = threadIdx.x;
  const int node0 = blockIdx.x * 16;
  const int r     = t >> 3;          // 0..15 : node row in tile
  const int g     = t & 7;           // 0..7  : 16-col group
  const int c0    = g * 16;
  const int node  = node0 + r;

  // ---- self features -> z[r][0..127] ----
  {
    const float4* sp = (const float4*)(feats + (size_t)node * F + c0);
    float4 s[4];
#pragma unroll
    for (int q = 0; q < 4; ++q) s[q] = sp[q];
    f16* zs = z + r * ZSTR + c0;
#pragma unroll
    for (int q = 0; q < 4; ++q) {
      zs[q * 4 + 0] = (f16)s[q].x; zs[q * 4 + 1] = (f16)s[q].y;
      zs[q * 4 + 2] = (f16)s[q].z; zs[q * 4 + 3] = (f16)s[q].w;
    }
  }
  // ---- neighbor mean -> z[r][128..255] ----
  {
    float acc[16];
#pragma unroll
    for (int i = 0; i < 16; ++i) acc[i] = 0.0f;
    const int* nb = neigh + (size_t)node * DEG;
    for (int e = 0; e < DEG; ++e) {
      int idx = nb[e];
      const float4* fp = (const float4*)(feats + (size_t)idx * F + c0);
#pragma unroll
      for (int q = 0; q < 4; ++q) {
        float4 v = fp[q];
        acc[q * 4 + 0] += v.x; acc[q * 4 + 1] += v.y;
        acc[q * 4 + 2] += v.z; acc[q * 4 + 3] += v.w;
      }
    }
    f16* zn = z + r * ZSTR + F + c0;
#pragma unroll
    for (int i = 0; i < 16; ++i) zn[i] = (f16)(acc[i] * 0.03125f);
  }
  __syncthreads();

  // ---- WMMA: [16 x 256] @ [256 x 128], wave w owns N-tiles {2w, 2w+1} ----
  const int lane = t & 31;
  const int wave = t >> 5;           // 0..3
  const int half = lane >> 4;        // K-half selector (A/B layouts)
  const int arow = lane & 15;        // A: matrix row / B,D: column-in-tile
  const f16* zrow = z + arow * ZSTR;
  const f16* wb0  = w1t + (size_t)((wave * 2 + 0) * 16 + arow) * KDIM;
  const f16* wb1  = w1t + (size_t)((wave * 2 + 1) * 16 + arow) * KDIM;

  v8f acc0 = {}, acc1 = {};
#pragma unroll
  for (int kc = 0; kc < 8; ++kc) {
    // A operand: lane holds row `arow`, 16 contiguous K at offset half*16
    v8h alo = *(const v8h*)(zrow + kc * 32 + half * 16);
    v8h ahi = *(const v8h*)(zrow + kc * 32 + half * 16 + 8);
    v16h a;
#pragma unroll
    for (int i = 0; i < 8; ++i) { a[i] = alo[i]; a[i + 8] = ahi[i]; }
    // B operand: lane holds column, 16 contiguous K at offset half*16
    v16h b0 = *(const v16h*)(wb0 + kc * 32 + half * 16);
    v16h b1 = *(const v16h*)(wb1 + kc * 32 + half * 16);
    acc0 = __builtin_amdgcn_wmma_f32_16x16x32_f16(false, a, false, b0,
                                                  (short)0, acc0, false, false);
    acc1 = __builtin_amdgcn_wmma_f32_16x16x32_f16(false, a, false, b1,
                                                  (short)0, acc1, false, false);
  }

  // ---- epilogue: +bias, ReLU, store f16 h1 (D: lane=col, vgpr v=row v+8*half)
#pragma unroll
  for (int nt = 0; nt < 2; ++nt) {
    int col = (wave * 2 + nt) * 16 + arow;
    float bias = b1[col];
    const v8f& c = nt ? acc1 : acc0;
#pragma unroll
    for (int v = 0; v < 8; ++v) {
      int row = v + 8 * half;
      float val = c[v] + bias;
      val = val > 0.0f ? val : 0.0f;
      h1[(size_t)(node0 + row) * F + col] = (f16)val;
    }
  }
}

// ---------------------------------------------------------------------------
// Layer 2: z2 = [h1 | mean(neigh h1)] ; out = z2 @ W2 + b2  (f32 out, 40 cols)
// ---------------------------------------------------------------------------
__global__ __launch_bounds__(128)
void gcn_layer2(const f16* __restrict__ h1, const int* __restrict__ neigh,
                const f16* __restrict__ w2t, const float* __restrict__ b2,
                float* __restrict__ out) {
  __shared__ f16 z[16 * ZSTR];
  const int t     = threadIdx.x;
  const int node0 = blockIdx.x * 16;
  const int r     = t >> 3;
  const int g     = t & 7;
  const int c0    = g * 16;
  const int node  = node0 + r;

  // ---- self embedding (already f16) -> z[r][0..127] ----
  {
    const v8h* sp = (const v8h*)(h1 + (size_t)node * F + c0);
    v8h s0 = sp[0], s1 = sp[1];
    *(v8h*)(z + r * ZSTR + c0)     = s0;
    *(v8h*)(z + r * ZSTR + c0 + 8) = s1;
  }
  // ---- neighbor mean (f32 accumulate of f16) -> z[r][128..255] ----
  {
    float acc[16];
#pragma unroll
    for (int i = 0; i < 16; ++i) acc[i] = 0.0f;
    const int* nb = neigh + (size_t)node * DEG;
    for (int e = 0; e < DEG; ++e) {
      int idx = nb[e];
      const v8h* hp = (const v8h*)(h1 + (size_t)idx * F + c0);
      v8h v0 = hp[0], v1 = hp[1];
#pragma unroll
      for (int i = 0; i < 8; ++i) { acc[i] += (float)v0[i]; acc[i + 8] += (float)v1[i]; }
    }
    f16* zn = z + r * ZSTR + F + c0;
#pragma unroll
    for (int i = 0; i < 16; ++i) zn[i] = (f16)(acc[i] * 0.03125f);
  }
  __syncthreads();

  // ---- WMMA: [16 x 256] @ [256 x 48(pad)], waves 0..2 own one N-tile ----
  const int lane = t & 31;
  const int wave = t >> 5;
  if (wave < 3) {                      // uniform per wave -> EXEC stays all-1s
    const int half = lane >> 4;
    const int arow = lane & 15;
    const f16* zrow = z + arow * ZSTR;
    const f16* wb   = w2t + (size_t)(wave * 16 + arow) * KDIM;

    v8f acc = {};
#pragma unroll
    for (int kc = 0; kc < 8; ++kc) {
      v8h alo = *(const v8h*)(zrow + kc * 32 + half * 16);
      v8h ahi = *(const v8h*)(zrow + kc * 32 + half * 16 + 8);
      v16h a;
#pragma unroll
      for (int i = 0; i < 8; ++i) { a[i] = alo[i]; a[i + 8] = ahi[i]; }
      v16h b = *(const v16h*)(wb + kc * 32 + half * 16);
      acc = __builtin_amdgcn_wmma_f32_16x16x32_f16(false, a, false, b,
                                                   (short)0, acc, false, false);
    }

    int col = wave * 16 + arow;
    if (col < NLAB) {
      float bias = b2[col];
#pragma unroll
      for (int v = 0; v < 8; ++v) {
        int row = v + 8 * half;
        out[(size_t)(node0 + row) * NLAB + col] = acc[v] + bias;
      }
    }
  }
}

// ---------------------------------------------------------------------------
extern "C" void kernel_launch(void* const* d_in, const int* in_sizes, int n_in,
                              void* d_out, int out_size, void* d_ws, size_t ws_size,
                              hipStream_t stream) {
  const float* feats = (const float*)d_in[0];
  const int*   neigh = (const int*)d_in[1];
  // d_in[2] = batch == arange(N): identity, unused
  const float* W1 = (const float*)d_in[3];
  const float* b1 = (const float*)d_in[4];
  const float* W2 = (const float*)d_in[5];
  const float* b2 = (const float*)d_in[6];
  float* out = (float*)d_out;

  char* ws = (char*)d_ws;
  f16* w1t = (f16*)ws;                                   //  65536 B
  f16* w2t = (f16*)(ws + 65536);                         //  24576 B
  f16* h1  = (f16*)(ws + 65536 + 24576);                 //  12.8 MB

  const int prep_elems = F * KDIM + NLABP * KDIM;        // 45056
  gcn_prep_weights<<<(prep_elems + 255) / 256, 256, 0, stream>>>(W1, W2, w1t, w2t);
  gcn_layer1<<<NNODES / 16, 128, 0, stream>>>(feats, neigh, w1t, b1, h1);
  gcn_layer2<<<NNODES / 16, 128, 0, stream>>>(h1, neigh, w2t, b2, out);
}
